// AttentionBahdanau_11716670784121
// MI455X (gfx1250) — matrix operand order repaired and checked by reading, not compiled, not run on hardware
//
#include <hip/hip_runtime.h>
#include <hip/hip_bf16.h>
#include <math.h>

// ---------------------------------------------------------------------------
// Bahdanau attention, fused for MI455X (gfx1250, wave32, WMMA).
// B=64, S=512, H=1024, AH=512.
//
// Main GEMM (enhy @ W_en^T, 68.7 GFLOP) runs on v_wmma_f32_16x16x32_bf16 with
// the tanh/W_out epilogue fused so the [B,S,H] intermediate never reaches
// memory. enhy is read twice total (GEMM pass + weighted-sum pass) = 256 MB
// -> ~11us HBM floor @ 23.3 TB/s; matrix pipe easily exceeds the 12.5 TF/s
// needed to stay memory-bound.
// ---------------------------------------------------------------------------

typedef __attribute__((ext_vector_type(16))) __bf16 v16bf;
typedef __attribute__((ext_vector_type(8)))  float  v8f;

union Frag16 {           // 16 bf16 = 32 bytes = two 16B chunks
    v16bf v;
    uint4 q[2];
};

#define BB  64
#define SS  512
#define HH  1024
#define AH  512
#define TS  128          // rows (b,s) per workgroup in scores kernel
#define LDA 1032         // padded bf16 LDS row stride (kills 16-way bank conflict)

__device__ __forceinline__ unsigned int f2bf1(float f) {
    unsigned int u = __float_as_uint(f);
    return (u + 0x7FFFu + ((u >> 16) & 1u)) >> 16;     // RNE
}

__device__ __forceinline__ unsigned int pack_bf16(float lo, float hi) {
#if __has_builtin(__builtin_amdgcn_cvt_pk_bf16_f32)
    typedef __attribute__((ext_vector_type(2))) __bf16 v2bf;
    v2bf p = __builtin_amdgcn_cvt_pk_bf16_f32(lo, hi);
    union { v2bf v; unsigned int u; } c;
    c.v = p;
    return c.u;
#else
    return f2bf1(lo) | (f2bf1(hi) << 16);
#endif
}

__device__ __forceinline__ float fast_tanh(float x) {
#if __has_builtin(__builtin_amdgcn_tanhf)
    return __builtin_amdgcn_tanhf(x);                  // v_tanh_f32 (gfx1250 trans op)
#else
    return tanhf(x);
#endif
}

// ---- kernel 1: W_en fp32 -> bf16 (row-major [o][h]) -----------------------
__global__ __launch_bounds__(256) void convw_kernel(const float* __restrict__ W_en,
                                                    unsigned short* __restrict__ Wbf) {
    int e = (blockIdx.x * 256 + threadIdx.x) * 4;
    float4 f = *(const float4*)(W_en + e);
    uint2 p;
    p.x = pack_bf16(f.x, f.y);
    p.y = pack_bf16(f.z, f.w);
    *(uint2*)(Wbf + e) = p;
}

// ---- kernel 2: colconst[b,o] = de_proj[b,o] + b_de[o] + b_en[o] -----------
__global__ __launch_bounds__(256) void colconst_kernel(const float* __restrict__ last_dehy,
                                                       const float* __restrict__ W_de,
                                                       const float* __restrict__ b_de,
                                                       const float* __restrict__ b_en,
                                                       float* __restrict__ colconst) {
    int b = blockIdx.y;
    int o = blockIdx.x * 256 + threadIdx.x;
    const float4* w = (const float4*)(W_de + (size_t)o * HH);
    const float4* x = (const float4*)(last_dehy + (size_t)b * HH);
    float acc = 0.f;
#pragma unroll 4
    for (int i = 0; i < HH / 4; ++i) {
        float4 ww = w[i], xx = x[i];
        acc += ww.x * xx.x + ww.y * xx.y + ww.z * xx.z + ww.w * xx.w;
    }
    colconst[(size_t)b * HH + o] = acc + b_de[o] + b_en[o];
}

// ---- kernel 3: fused WMMA GEMM + tanh + W_out reduction -> scores[b,s] ----
// grid (S/TS=4, B), 256 threads (8 waves). Wave w owns columns [w*128,(w+1)*128).
__global__ __launch_bounds__(256) void scores_kernel(const float* __restrict__ enhy,
                                                     const float* __restrict__ past_attn,
                                                     const float* __restrict__ colconst,
                                                     const unsigned short* __restrict__ Wbf,
                                                     const float* __restrict__ W_cv,
                                                     const float* __restrict__ b_cv,
                                                     const float* __restrict__ W_out,
                                                     float* __restrict__ scores) {
    extern __shared__ char smemraw[];
    unsigned short* sA   = (unsigned short*)smemraw;          // [TS][LDA] bf16
    float*          sPA  = (float*)(smemraw + TS * LDA * 2);  // past_attn tile
    float*          sSc  = sPA + TS;                          // score accum

    const int b   = blockIdx.y;
    const int s0  = blockIdx.x * TS;
    const int tid = threadIdx.x;

    // Stage A tile: fp32 global -> bf16 LDS (coalesced 16B loads, one row/iter)
    const float* gA = enhy + ((size_t)b * SS + s0) * HH;
#pragma unroll 2
    for (int row = 0; row < TS; ++row) {
        float4 f = *(const float4*)(gA + (size_t)row * HH + tid * 4);
        uint2 p;
        p.x = pack_bf16(f.x, f.y);
        p.y = pack_bf16(f.z, f.w);
        *(uint2*)(sA + row * LDA + tid * 4) = p;
    }
    if (tid < TS) {
        sPA[tid] = past_attn[(size_t)b * SS + s0 + tid];
        sSc[tid] = 0.f;
    }
    __syncthreads();

    const int wave   = tid >> 5;
    const int lane   = tid & 31;
    const int lane15 = lane & 15;
    const int laneHi = lane >> 4;
    const int obase  = wave * 128;   // this wave's 128-column slice of o

    // per-lane A base: row = mt*16 + lane15, K chunks at +laneHi*8 / +16+laneHi*8
    const unsigned short* Ap = sA + lane15 * LDA + laneHi * 8;

    for (int nc = 0; nc < 8; ++nc) {
        const int o = obase + nc * 16 + lane15;           // this lane's column
        const unsigned short* Bp = Wbf + (size_t)o * HH + laneHi * 8;

        v8f C[8];
#pragma unroll
        for (int m = 0; m < 8; ++m)
            C[m] = (v8f){0.f, 0.f, 0.f, 0.f, 0.f, 0.f, 0.f, 0.f};

#pragma unroll 2
        for (int kb = 0; kb < 8; ++kb) {                  // K blocks of 128; unroll 2
            Frag16 Bf[4];                                 // -> 2 B register sets, loads
#pragma unroll                                            //    of kb+1 hoist over kb
            for (int ks = 0; ks < 4; ++ks) {              // B: 8 bf16 at +laneHi*8, 8 at +16
                const uint4* p = (const uint4*)(Bp + kb * 128 + ks * 32);
                Bf[ks].q[0] = p[0];
                Bf[ks].q[1] = p[2];
            }
            if (kb < 7)                                   // global_prefetch next K-block
                __builtin_prefetch((const void*)(Bp + (kb + 1) * 128), 0, 1);

            // Flattened (ks,mt) with 2-deep A-fragment double buffer: WMMA step i
            // waits only for loads issued at step i-1 (dscnt<=2, not 0), and DS
            // writes land >=2 WMMAs after their registers were last read.
            Frag16 Af[2];
            {
                const unsigned short* ap0 = Ap + kb * 128;
                Af[0].q[0] = *(const uint4*)(ap0);
                Af[0].q[1] = *(const uint4*)(ap0 + 16);
            }
#pragma unroll
            for (int idx = 0; idx < 32; ++idx) {
                const int ks  = idx >> 3;                 // mt inner: 8 independent
                const int mt  = idx & 7;                  // accumulators between reuses
                const int nid = (idx + 1) & 31;           // last step reloads idx0 (dummy)
                const unsigned short* ap =
                    Ap + (nid & 7) * 16 * LDA + kb * 128 + (nid >> 3) * 32;
                Af[(idx + 1) & 1].q[0] = *(const uint4*)(ap);       // K k..k+7  (+laneHi*8)
                Af[(idx + 1) & 1].q[1] = *(const uint4*)(ap + 16);  // K k+16..23(+laneHi*8)
                C[mt] = __builtin_amdgcn_wmma_f32_16x16x32_bf16(
                    false, Af[idx & 1].v, false, Bf[ks].v, (short)0, C[mt], false, false);
            }
        }

        // fused epilogue: tanh(agg) * W_out, reduced over this 16-col chunk
        const float cc  = colconst[(size_t)b * HH + o];
        const float wcv = W_cv[o];
        const float bcv = b_cv[o];
        const float wo  = W_out[o];
#pragma unroll
        for (int mt = 0; mt < 8; ++mt) {
#pragma unroll
            for (int r = 0; r < 8; ++r) {
                const int row = mt * 16 + laneHi * 8 + r; // C layout: lanes<16 M=r, >=16 M=r+8
                float val = fast_tanh(C[mt][r] + cc + sPA[row] * wcv + bcv) * wo;
                val += __shfl_xor(val, 1);
                val += __shfl_xor(val, 2);
                val += __shfl_xor(val, 4);
                val += __shfl_xor(val, 8);
                if (lane15 == 0) atomicAdd(&sSc[row], val);   // ds_add_f32
            }
        }
    }
    __syncthreads();
    if (tid < TS) scores[(size_t)b * SS + s0 + tid] = sSc[tid];
}

// ---- kernel 4: softmax over S per batch + hidden_attn passthrough ---------
__global__ __launch_bounds__(512) void softmax_kernel(const float* __restrict__ scores,
                                                      const float* __restrict__ hidden,
                                                      float* __restrict__ out) {
    __shared__ float red[SS];
    const int b = blockIdx.x, tid = threadIdx.x;
    float v = scores[(size_t)b * SS + tid];
    red[tid] = v;
    __syncthreads();
    for (int off = 256; off > 0; off >>= 1) {
        if (tid < off) red[tid] = fmaxf(red[tid], red[tid + off]);
        __syncthreads();
    }
    const float m = red[0];
    __syncthreads();
    float e = expf(v - m);
    red[tid] = e;
    __syncthreads();
    for (int off = 256; off > 0; off >>= 1) {
        if (tid < off) red[tid] += red[tid + off];
        __syncthreads();
    }
    const float a = e / red[0];
    out[BB * HH + (size_t)b * SS + tid]           = a;                            // attn
    out[BB * HH + BB * SS + (size_t)b * AH + tid] = hidden[(size_t)b * AH + tid]; // passthrough
}

// ---- kernel 5: h_attn[b,h] = sum_s attn[b,s] * enhy[b,s,h] ----------------
__global__ __launch_bounds__(256) void wsum_kernel(const float* __restrict__ attn,
                                                   const float* __restrict__ enhy,
                                                   float* __restrict__ out_h) {
    const int b   = blockIdx.x;
    const int col = blockIdx.y * 256 + threadIdx.x;
    const float* e = enhy + (size_t)b * SS * HH + col;
    const float* a = attn + (size_t)b * SS;
    float acc = 0.f;
#pragma unroll 8
    for (int s = 0; s < SS; ++s)
        acc = fmaf(a[s], e[(size_t)s * HH], acc);
    out_h[(size_t)b * HH + col] = acc;
}

// ---------------------------------------------------------------------------
extern "C" void kernel_launch(void* const* d_in, const int* in_sizes, int n_in,
                              void* d_out, int out_size, void* d_ws, size_t ws_size,
                              hipStream_t stream) {
    const float* last_dehy = (const float*)d_in[0];
    const float* enhy      = (const float*)d_in[1];
    const float* past_attn = (const float*)d_in[2];
    const float* hidden    = (const float*)d_in[3];
    const float* W_en      = (const float*)d_in[4];
    const float* b_en      = (const float*)d_in[5];
    const float* W_de      = (const float*)d_in[6];
    const float* b_de      = (const float*)d_in[7];
    const float* W_cv      = (const float*)d_in[8];
    const float* b_cv      = (const float*)d_in[9];
    const float* W_out     = (const float*)d_in[10];
    float* out = (float*)d_out;

    char* ws = (char*)d_ws;
    unsigned short* Wbf    = (unsigned short*)ws;                       // 2 MB
    float* colconst        = (float*)(ws + (size_t)2 * 1024 * 1024);    // 256 KB
    float* scores          = (float*)(ws + (size_t)2 * 1024 * 1024 + BB * HH * 4);

    // 1) W_en -> bf16
    convw_kernel<<<(HH * HH) / (256 * 4), 256, 0, stream>>>(W_en, Wbf);
    // 2) de_proj + biases (per-(b,o) column constant)
    colconst_kernel<<<dim3(HH / 256, BB), 256, 0, stream>>>(last_dehy, W_de, b_de, b_en, colconst);
    // 3) WMMA scores (258 KB dynamic LDS: A tile + past_attn + score accum)
    size_t shbytes = (size_t)TS * LDA * 2 + (size_t)TS * 4 * 2;
    scores_kernel<<<dim3(SS / TS, BB), 256, shbytes, stream>>>(
        enhy, past_attn, colconst, Wbf, W_cv, b_cv, W_out, scores);
    // 4) softmax + hidden passthrough (writes attn section of out)
    softmax_kernel<<<BB, SS, 0, stream>>>(scores, hidden, out);
    // 5) attention-weighted context vector
    wsum_kernel<<<dim3(BB, HH / 256), 256, 0, stream>>>(out + BB * HH, enhy, out);
}
